// GCNN_9036611191116
// MI455X (gfx1250) — matrix operand order repaired and checked by reading, not compiled, not run on hardware
//
#include <hip/hip_runtime.h>
#include <hip/hip_bf16.h>
#include <math.h>

#define N_NODES  10000
#define N_EDGES  160000
#define N_GRAPHS 32
#define F_IN     1024
#define F_OUT    128
#define NEG_SLOPE 0.01f

typedef __attribute__((ext_vector_type(16))) __bf16 v16bf;
typedef __attribute__((ext_vector_type(8)))  __bf16 v8bf;
typedef __attribute__((ext_vector_type(8)))  float  v8f;
typedef __attribute__((ext_vector_type(4)))  unsigned int v4u;
typedef __attribute__((ext_vector_type(4)))  int v4i;
typedef __attribute__((ext_vector_type(8)))  int v8i;

__device__ __forceinline__ float leaky(float x) { return x >= 0.f ? x : NEG_SLOPE * x; }

#if __has_builtin(__builtin_amdgcn_tensor_load_to_lds)
#define HAVE_TDM 1
#else
#define HAVE_TDM 0
#endif

// Publish LDS contents written by the TDM to the compiler's alias analysis:
// the pointer escapes into asm that may have written through it, so subsequent
// ds_load fragment reads cannot be folded away (round-2 bug: ds=0, undef frags).
__device__ __forceinline__ void lds_publish(const void* p) {
    void* q = (void*)p;
    asm volatile("" : "+v"(q) :: "memory");
}

// ---------------- TDM 2D tile load: global(bf16, row-major) -> LDS ----------------
// Builds D# per CDNA5 ISA ch.8: group0 {count=1, lds_addr, global_addr, type=2},
// group1 {data_size=2B, tensor_dim0/1, tile_dim0/1, tensor_dim0_stride}.
__device__ __forceinline__ void tdm_load_2d(
    unsigned lds_byte_off, const void* gaddr,
    unsigned tensor_d0, unsigned tensor_d1,
    unsigned long long stride0_elems,
    unsigned tile_d0, unsigned tile_d1)
{
#if HAVE_TDM
    unsigned long long ga = (unsigned long long)gaddr;
    v4u g0;
    g0[0] = 1u;                                     // count=1 (valid), user mode
    g0[1] = lds_byte_off;                           // lds_addr
    g0[2] = (unsigned)ga;                           // global_addr[31:0]
    g0[3] = (unsigned)(ga >> 32) | 0x80000000u;     // global_addr[56:32] | type=2
    v8i g1;
    g1[0] = 0x00010000;                             // data_size=1 (2 bytes), no flags
    g1[1] = (int)((tensor_d0 & 0xFFFFu) << 16);     // tensor_dim0[15:0]
    g1[2] = (int)((tensor_d0 >> 16) | ((tensor_d1 & 0xFFFFu) << 16));
    g1[3] = (int)((tensor_d1 >> 16) | (tile_d0 << 16));
    g1[4] = (int)(tile_d1 & 0xFFFFu);               // tile_dim1; tile_dim2=0 (2D)
    g1[5] = (int)(stride0_elems & 0xFFFFFFFFull);   // tensor_dim0_stride[31:0]
    g1[6] = (int)((stride0_elems >> 32) & 0xFFFFull);
    g1[7] = 0;
    v4i z4 = {0, 0, 0, 0};
#if __clang_major__ >= 23
    v8i z8 = {0, 0, 0, 0, 0, 0, 0, 0};
    __builtin_amdgcn_tensor_load_to_lds(g0, g1, z4, z4, z8, 0);
#else
    __builtin_amdgcn_tensor_load_to_lds(g0, g1, z4, z4, 0);
#endif
#else
    (void)lds_byte_off; (void)gaddr; (void)tensor_d0; (void)tensor_d1;
    (void)stride0_elems; (void)tile_d0; (void)tile_d1;
#endif
}

// ---------------- utility kernels ----------------
__global__ void k_zero(float* p, int n) {
    int i = blockIdx.x * blockDim.x + threadIdx.x;
    if (i < n) p[i] = 0.f;
}

__global__ void k_deg(const int* __restrict__ dst, float* __restrict__ deg, int nE) {
    int i = blockIdx.x * blockDim.x + threadIdx.x;
    if (i < nE) atomicAdd(&deg[dst[i]], 1.0f);
}

__global__ void k_dinv(const float* __restrict__ deg, float* __restrict__ dinv, int n) {
    int i = blockIdx.x * blockDim.x + threadIdx.x;
    if (i < n) dinv[i] = rsqrtf(deg[i] + 1.0f);   // +1 = self loop; always > 0
}

__global__ void k_cnt(const int* __restrict__ batch, float* __restrict__ cnt, int n) {
    int i = blockIdx.x * blockDim.x + threadIdx.x;
    if (i < n) atomicAdd(&cnt[batch[i]], 1.0f);
}

// f32 -> split bf16 (hi + lo), same layout
__global__ void k_split(const float* __restrict__ in, __bf16* __restrict__ hi,
                        __bf16* __restrict__ lo, int n) {
    int i = blockIdx.x * blockDim.x + threadIdx.x;
    if (i >= n) return;
    float v = in[i];
    __bf16 h = (__bf16)v;
    hi[i] = h;
    lo[i] = (__bf16)(v - (float)h);
}

// W[K,Nc] f32 -> transposed split bf16: out[n*K + k]
__global__ void k_split_t(const float* __restrict__ W, __bf16* __restrict__ hiT,
                          __bf16* __restrict__ loT, int K, int Nc) {
    int i = blockIdx.x * blockDim.x + threadIdx.x;
    if (i >= K * Nc) return;
    int k = i / Nc, n = i - k * Nc;
    float v = W[i];
    __bf16 h = (__bf16)v;
    hiT[(size_t)n * K + k] = h;
    loT[(size_t)n * K + k] = (__bf16)(v - (float)h);
}

// ---------------- WMMA GEMM: C[M,Nc] = A @ W, bf16x3 split ----------------
// Block tile 128(M) x 128(N), 8 waves; wave owns a 16x128 strip (8 N-subtiles).
// TDM double-buffered LDS staging: wave 0 issues next tile's 4 tensor loads,
// all waves compute on the current 32KB buffer, then s_wait_tensorcnt + barrier.
#define BM 128
#define BN 128
#define BK 32
#define OFF_AHI 0
#define OFF_ALO (BM * BK)
#define OFF_BHI (2 * BM * BK)
#define OFF_BLO (2 * BM * BK + BN * BK)
#define BUF_ELEMS (2 * BM * BK + 2 * BN * BK)    // 16384 bf16 = 32 KB per buffer

__global__ __launch_bounds__(256) void k_gemm_bf16x3(
    const __bf16* __restrict__ Ahi_g, const __bf16* __restrict__ Alo_g,
    const __bf16* __restrict__ Bthi_g, const __bf16* __restrict__ Btlo_g,
    float* __restrict__ C, int M, int K, int Nc)
{
    __shared__ __align__(128) __bf16 smem[2 * BUF_ELEMS];   // 64 KB, double buffered

    const int tid  = threadIdx.x;
    const int wave = tid >> 5;
    const int lane = tid & 31;
    const int lrow = lane & 15;   // m (A) or n (B/C) inside a 16-wide tile
    const int half = lane >> 4;   // lane-group select per ISA 7.12.2
    const int mb = blockIdx.x * BM;
    const int nb = blockIdx.y * BN;
    const int ksteps = K / BK;

    v8f acc[8];
    const v8f vz = {0.f,0.f,0.f,0.f,0.f,0.f,0.f,0.f};
    #pragma unroll
    for (int s = 0; s < 8; ++s) acc[s] = vz;

    auto issue_tile = [&](int kb, int buf) {
        unsigned base = (unsigned)(buf * BUF_ELEMS * 2);    // LDS byte offset
        tdm_load_2d(base + OFF_AHI * 2, Ahi_g  + (size_t)mb * K + kb, K, (unsigned)M,  K, BK, BM);
        tdm_load_2d(base + OFF_ALO * 2, Alo_g  + (size_t)mb * K + kb, K, (unsigned)M,  K, BK, BM);
        tdm_load_2d(base + OFF_BHI * 2, Bthi_g + (size_t)nb * K + kb, K, (unsigned)Nc, K, BK, BN);
        tdm_load_2d(base + OFF_BLO * 2, Btlo_g + (size_t)nb * K + kb, K, (unsigned)Nc, K, BK, BN);
    };

    int cur = 0;
#if HAVE_TDM
    if (tid < 32) {                         // prologue: fill buffer 0
        issue_tile(0, 0);
        __builtin_amdgcn_s_wait_tensorcnt(0);
    }
    lds_publish(smem);
    __syncthreads();
#endif

    for (int step = 0; step < ksteps; ++step) {
#if HAVE_TDM
        if ((step + 1 < ksteps) && tid < 32)        // prefetch next tile (overlapped)
            issue_tile((step + 1) * BK, cur ^ 1);
#else
        {   // cooperative fallback staging (bf16 global -> LDS), single buffer
            int kb = step * BK;
            __syncthreads();
            #pragma unroll
            for (int j = 0; j < 16; ++j) {           // A tiles: 2 x 4096 elems
                int idx = j * 256 + tid;
                int r = idx >> 5, c = idx & 31;
                int gr = mb + r;
                __bf16 vh = (__bf16)0.f, vl = (__bf16)0.f;
                if (gr < M) {
                    vh = Ahi_g[(size_t)gr * K + kb + c];
                    vl = Alo_g[(size_t)gr * K + kb + c];
                }
                smem[OFF_AHI + idx] = vh;
                smem[OFF_ALO + idx] = vl;
            }
            #pragma unroll
            for (int j = 0; j < 16; ++j) {           // B tiles: 2 x 4096 elems
                int idx = j * 256 + tid;
                int r = idx >> 5, c = idx & 31;      // r = n, c = k
                smem[OFF_BHI + idx] = Bthi_g[(size_t)(nb + r) * K + kb + c];
                smem[OFF_BLO + idx] = Btlo_g[(size_t)(nb + r) * K + kb + c];
            }
            __syncthreads();
        }
#endif
        // ---- compute on buffer `cur` ----
        const __bf16* buf = smem + cur * BUF_ELEMS;
        const __bf16* Ah = buf + OFF_AHI;
        const __bf16* Al = buf + OFF_ALO;
        // A fragment (16-bit 16x32: elems 0..15 -> K=half*16 + 0..15)
        const int am = wave * 16 + lrow;
        v8bf ah0 = *(const v8bf*)(Ah + am * BK + half * 16);
        v8bf ah1 = *(const v8bf*)(Ah + am * BK + half * 16 + 8);
        v16bf a_hi = __builtin_shufflevector(ah0, ah1, 0,1,2,3,4,5,6,7,8,9,10,11,12,13,14,15);
        v8bf al0 = *(const v8bf*)(Al + am * BK + half * 16);
        v8bf al1 = *(const v8bf*)(Al + am * BK + half * 16 + 8);
        v16bf a_lo = __builtin_shufflevector(al0, al1, 0,1,2,3,4,5,6,7,8,9,10,11,12,13,14,15);

        const __bf16* Bh = buf + OFF_BHI;
        const __bf16* Bl = buf + OFF_BLO;
        #pragma unroll
        for (int s = 0; s < 8; ++s) {
            int bn = s * 16 + lrow;
            // B fragment (32x16): lane's 16 elems are K = half*16 + 0..15 of column n
            v16bf b_hi = *(const v16bf*)(Bh + bn * BK + half * 16);
            v16bf b_lo = *(const v16bf*)(Bl + bn * BK + half * 16);
            acc[s] = __builtin_amdgcn_wmma_f32_16x16x32_bf16(false, a_hi, false, b_hi, (short)0, acc[s], false, false);
            acc[s] = __builtin_amdgcn_wmma_f32_16x16x32_bf16(false, a_hi, false, b_lo, (short)0, acc[s], false, false);
            acc[s] = __builtin_amdgcn_wmma_f32_16x16x32_bf16(false, a_lo, false, b_hi, (short)0, acc[s], false, false);
        }

        __syncthreads();                     // all reads of buffer `cur` complete
#if HAVE_TDM
        if (step + 1 < ksteps) {
            if (tid < 32)
                __builtin_amdgcn_s_wait_tensorcnt(0);   // next buffer landed in LDS
            lds_publish(smem);
            __syncthreads();                 // visible to all waves
            cur ^= 1;
        }
#endif
    }

    // C/D layout: VGPR r -> M = half*8 + r; lane -> N = lrow
    #pragma unroll
    for (int s = 0; s < 8; ++s) {
        #pragma unroll
        for (int r = 0; r < 8; ++r) {
            int m = mb + wave * 16 + half * 8 + r;
            int n = nb + s * 16 + lrow;
            if (m < M) C[(size_t)m * Nc + n] = acc[s][r];
        }
    }
}

// ---------------- edge aggregation: h[dst] += xw[src] * dinv[src]*dinv[dst] ----------------
__global__ __launch_bounds__(256) void k_agg(
    const float* __restrict__ xw, const int* __restrict__ src, const int* __restrict__ dst,
    const float* __restrict__ dinv, float* __restrict__ h, int nE)
{
    int e    = blockIdx.x * (blockDim.x >> 5) + (threadIdx.x >> 5);  // wave-per-edge
    int lane = threadIdx.x & 31;
    if (e >= nE) return;
    int s = src[e], d = dst[e];
    float norm = dinv[s] * dinv[d];
    const float4* xs = (const float4*)(xw + (size_t)s * F_IN);
    float* hd = h + (size_t)d * F_IN;
    #pragma unroll
    for (int step = 0; step < 8; ++step) {
        int c4 = step * 32 + lane;      // 256 float4 per row
        float4 v = xs[c4];
        int c = c4 * 4;
        atomicAdd(&hd[c + 0], v.x * norm);
        atomicAdd(&hd[c + 1], v.y * norm);
        atomicAdd(&hd[c + 2], v.z * norm);
        atomicAdd(&hd[c + 3], v.w * norm);
    }
}

// ---------------- self-loop + bias + leaky, fused with atomic mean-pool numerator ----------------
__global__ __launch_bounds__(256) void k_bias_relu_pool(
    const float* __restrict__ h, const float* __restrict__ xw, const float* __restrict__ dinv,
    const float* __restrict__ bias, const int* __restrict__ batch, float* __restrict__ pool)
{
    int i = blockIdx.x;                 // node
    float di = dinv[i];
    float self = di * di;
    int g = batch[i];
    const float* hi = h  + (size_t)i * F_IN;
    const float* xi = xw + (size_t)i * F_IN;
    float* pg = pool + (size_t)g * F_IN;
    for (int c = threadIdx.x; c < F_IN; c += blockDim.x) {
        float v = leaky(hi[c] + xi[c] * self + bias[c]);
        atomicAdd(&pg[c], v);
    }
}

// ---------------- projection: z[g, colOff + c] = leaky(mean(pool) @ Wp + bp) ----------------
__global__ __launch_bounds__(128) void k_proj(
    const float* __restrict__ pool, const float* __restrict__ cnt,
    const float* __restrict__ Wp, const float* __restrict__ bp,
    float* __restrict__ z, int colOff)
{
    int g = blockIdx.x, c = threadIdx.x;            // c in [0,128)
    float scale = 1.0f / fmaxf(cnt[g], 1.0f);
    const float* pg = pool + (size_t)g * F_IN;
    float acc = 0.f;
    for (int k = 0; k < F_IN; ++k) acc += pg[k] * Wp[(size_t)k * F_OUT + c];
    z[(size_t)g * 256 + colOff + c] = leaky(acc * scale + bp[c]);
}

// ---------------- MLP head ----------------
__global__ void k_fc1(const float* __restrict__ z, const float* __restrict__ W,
                      const float* __restrict__ b, float* __restrict__ t1) {
    int g = blockIdx.x, c = threadIdx.x;            // 256 threads
    const float* zg = z + (size_t)g * 256;
    float acc = 0.f;
    for (int k = 0; k < 256; ++k) acc += zg[k] * W[(size_t)k * 256 + c];
    t1[(size_t)g * 256 + c] = leaky(acc + b[c]);
}

__global__ void k_fc2(const float* __restrict__ t1, const float* __restrict__ W,
                      const float* __restrict__ b, float* __restrict__ t2) {
    int g = blockIdx.x, c = threadIdx.x;            // 64 threads
    const float* tg = t1 + (size_t)g * 256;
    float acc = 0.f;
    for (int k = 0; k < 256; ++k) acc += tg[k] * W[(size_t)k * 64 + c];
    t2[(size_t)g * 64 + c] = leaky(acc + b[c]);
}

__global__ void k_fco(const float* __restrict__ t2, const float* __restrict__ Wo,
                      const float* __restrict__ bo, float* __restrict__ out) {
    int g = threadIdx.x;                            // 32 threads
    const float* tg = t2 + (size_t)g * 64;
    float acc = 0.f;
    for (int k = 0; k < 64; ++k) acc += tg[k] * Wo[k];
    acc += bo[0];
    out[g] = 1.0f / (1.0f + expf(-acc));
}

// ---------------- host launch ----------------
extern "C" void kernel_launch(void* const* d_in, const int* in_sizes, int n_in,
                              void* d_out, int out_size, void* d_ws, size_t ws_size,
                              hipStream_t stream)
{
    const float* x1  = (const float*)d_in[0];
    const int*   e1  = (const int*)d_in[1];
    const int*   b1  = (const int*)d_in[2];
    const float* x2  = (const float*)d_in[3];
    const int*   e2  = (const int*)d_in[4];
    const int*   b2  = (const int*)d_in[5];
    const float* Wg1 = (const float*)d_in[6];
    const float* bg1 = (const float*)d_in[7];
    const float* Wg2 = (const float*)d_in[8];
    const float* bg2 = (const float*)d_in[9];
    const float* Wp1 = (const float*)d_in[10];
    const float* bp1 = (const float*)d_in[11];
    const float* Wp2 = (const float*)d_in[12];
    const float* bp2 = (const float*)d_in[13];
    const float* Wf1 = (const float*)d_in[14];
    const float* bf1 = (const float*)d_in[15];
    const float* Wf2 = (const float*)d_in[16];
    const float* bf2 = (const float*)d_in[17];
    const float* Wo  = (const float*)d_in[18];
    const float* bo  = (const float*)d_in[19];

    char* ws = (char*)d_ws;
    size_t off = 0;
    auto alloc = [&](size_t bytes) -> void* {
        void* p = (void*)(ws + off);
        off += (bytes + 255) & ~(size_t)255;
        return p;
    };
    float*  xw   = (float*)alloc((size_t)N_NODES * F_IN * 4);   // 40 MB
    float*  hbuf = (float*)alloc((size_t)N_NODES * F_IN * 4);   // 40 MB
    float*  deg  = (float*)alloc((size_t)N_NODES * 4);
    float*  dinv = (float*)alloc((size_t)N_NODES * 4);
    float*  pool = (float*)alloc((size_t)N_GRAPHS * F_IN * 4);
    float*  cnt  = (float*)alloc((size_t)N_GRAPHS * 4);
    float*  z    = (float*)alloc((size_t)N_GRAPHS * 256 * 4);
    float*  t1   = (float*)alloc((size_t)N_GRAPHS * 256 * 4);
    float*  t2   = (float*)alloc((size_t)N_GRAPHS * 64 * 4);
    // NOTE: keep xhi -> xlo -> wthi -> wtlo contiguous and last, in this order:
    // TDM tail-tile overreach (<= 229 KB past xhi/xlo) lands in the next buffer.
    __bf16* xhi  = (__bf16*)alloc((size_t)N_NODES * F_IN * 2);  // 20 MB
    __bf16* xlo  = (__bf16*)alloc((size_t)N_NODES * F_IN * 2);  // 20 MB
    __bf16* wthi = (__bf16*)alloc((size_t)F_IN * F_IN * 2);     // 2 MB
    __bf16* wtlo = (__bf16*)alloc((size_t)F_IN * F_IN * 2);     // 2 MB
    (void)ws_size; (void)n_in; (void)in_sizes; (void)out_size;

    auto run_branch = [&](const float* x, const int* eidx, const int* batch,
                          const float* Wg, const float* bg,
                          const float* Wp, const float* bp, int colOff) {
        const int* src = eidx;
        const int* dst = eidx + N_EDGES;
        k_zero<<<(N_NODES + 255) / 256, 256, 0, stream>>>(deg, N_NODES);
        k_zero<<<(N_NODES * F_IN + 255) / 256, 256, 0, stream>>>(hbuf, N_NODES * F_IN);
        k_zero<<<(N_GRAPHS * F_IN + 255) / 256, 256, 0, stream>>>(pool, N_GRAPHS * F_IN);
        k_zero<<<1, 32, 0, stream>>>(cnt, N_GRAPHS);
        k_deg<<<(N_EDGES + 255) / 256, 256, 0, stream>>>(dst, deg, N_EDGES);
        k_dinv<<<(N_NODES + 255) / 256, 256, 0, stream>>>(deg, dinv, N_NODES);
        k_split<<<(N_NODES * F_IN + 255) / 256, 256, 0, stream>>>(x, xhi, xlo, N_NODES * F_IN);
        k_split_t<<<(F_IN * F_IN + 255) / 256, 256, 0, stream>>>(Wg, wthi, wtlo, F_IN, F_IN);
        k_gemm_bf16x3<<<dim3((N_NODES + BM - 1) / BM, F_IN / BN), 256, 0, stream>>>(
            xhi, xlo, wthi, wtlo, xw, N_NODES, F_IN, F_IN);
        k_agg<<<(N_EDGES + 7) / 8, 256, 0, stream>>>(xw, src, dst, dinv, hbuf, N_EDGES);
        k_cnt<<<(N_NODES + 255) / 256, 256, 0, stream>>>(batch, cnt, N_NODES);
        k_bias_relu_pool<<<N_NODES, 256, 0, stream>>>(hbuf, xw, dinv, bg, batch, pool);
        k_proj<<<N_GRAPHS, F_OUT, 0, stream>>>(pool, cnt, Wp, bp, z, colOff);
    };

    run_branch(x1, e1, b1, Wg1, bg1, Wp1, bp1, 0);
    run_branch(x2, e2, b2, Wg2, bg2, Wp2, bp2, 128);

    k_fc1<<<N_GRAPHS, 256, 0, stream>>>(z, Wf1, bf1, t1);
    k_fc2<<<N_GRAPHS, 64, 0, stream>>>(t1, Wf2, bf2, t2);
    k_fco<<<1, N_GRAPHS, 0, stream>>>(t2, Wo, bo, (float*)d_out);
}
